// SelfAttentionHead_5712306503719
// MI455X (gfx1250) — matrix operand order repaired and checked by reading, not compiled, not run on hardware
//
#include <hip/hip_runtime.h>

// ---------------------------------------------------------------------------
// Self-attention head (B=4, T=4096, C=768, H=64), causal, k@q^T variant.
//   Stage 1: templated K/Q/V projection, v_wmma_f32_16x16x32_f16, 4 M-tiles
//            per wave (B-fragment reuse), bias fused, K pre-scaled by
//            C^-0.5*log2(e) (exp2-domain softmax), V transposed via LDS.
//   Stage 2: split-s flash attention: 2 waves per 16-row tile (even/odd
//            32-col chunks), mask-free fast path below the diagonal,
//            private online-softmax state merged through LDS at the end.
// ---------------------------------------------------------------------------

#define TOKENS 4096
#define BATCH  4
#define EMB    768
#define HEAD   64
#define BT     (BATCH * TOKENS)
#define NEGINF (-3.0e38f)
// 768^-0.5 * log2(e): softmax computed with exp2
#define KSCALE 0.052058784f

typedef __attribute__((ext_vector_type(16))) _Float16 v16h;
typedef __attribute__((ext_vector_type(8)))  float    v8f;
typedef __attribute__((ext_vector_type(4)))  float    v4f;
typedef __attribute__((ext_vector_type(4)))  int      v4i;

union Frag16 { v16h h; v4i q[2]; };

__device__ __forceinline__ v16h load_frag16(const _Float16* p0, const _Float16* p1) {
    Frag16 f;
    f.q[0] = *(const v4i*)p0;
    f.q[1] = *(const v4i*)p1;
    return f.h;
}

// Compiler-only ordering fence for the wave-private LDS round-trip
// (LDS ops are issued/completed in order within a wave on CDNA5).
__device__ __forceinline__ void lds_order_fence() {
    __builtin_amdgcn_sched_barrier(0);
    asm volatile("" ::: "memory");
}

// ---------------------------------------------------------------------------
// Stage 1: projection.  MAT: 0=K (pre-scaled), 1=Q, 2=V (stored transposed).
// grid = (BT/64, 4); one wave computes a 64x16 strip: 4 row-tiles sharing
// each weight B-fragment; K-loop over 768 in steps of 32.
// ---------------------------------------------------------------------------
template <int MAT>
__global__ __launch_bounds__(32) void proj_kernel(
    const float* __restrict__ x, const float* __restrict__ W,
    const float* __restrict__ bias, _Float16* __restrict__ dst)
{
    __shared__ _Float16 tile[16 * 16];   // V transpose staging (unused for K/Q)

    const int lane = threadIdx.x;
    const int lh   = lane >> 4;          // lane half: 0 or 1
    const int lid  = lane & 15;
    const int m0   = blockIdx.x * 64;
    const int n0   = blockIdx.y * 16;

    const float* xr0 = x + (size_t)(m0 + lid) * EMB + lh * 8;
    const int bn = n0 + lid;

    v8f acc[4] = {{}, {}, {}, {}};
    for (int k0 = 0; k0 < EMB; k0 += 32) {
        // B 32x16 fragment: element j = W[k0 + lh*8 + j + (j>=8 ? 8 : 0)][bn] (coalesced).
        const float* wp = W + (size_t)(k0 + lh * 8) * HEAD + bn;
        v16h Bf;
#pragma unroll
        for (int j = 0; j < 16; ++j) Bf[j] = (_Float16)wp[(size_t)(j + (j >= 8 ? 8 : 0)) * HEAD];

#pragma unroll
        for (int mt = 0; mt < 4; ++mt) {
            const float* xr = xr0 + (size_t)mt * 16 * EMB + k0;
            v4f a0 = *(const v4f*)(xr);
            v4f a1 = *(const v4f*)(xr + 4);
            v4f a2 = *(const v4f*)(xr + 16);
            v4f a3 = *(const v4f*)(xr + 20);
            v16h A;
#pragma unroll
            for (int i = 0; i < 4; ++i) {
                A[i]      = (_Float16)a0[i];
                A[4 + i]  = (_Float16)a1[i];
                A[8 + i]  = (_Float16)a2[i];
                A[12 + i] = (_Float16)a3[i];
            }
            acc[mt] = __builtin_amdgcn_wmma_f32_16x16x32_f16(
                          false, A, false, Bf, (short)0, acc[mt], false, false);
        }
    }

    const float bb = bias[bn];
#pragma unroll
    for (int mt = 0; mt < 4; ++mt) {
        const int mbase = m0 + mt * 16;
        if (MAT < 2) {
            // Row-major f16 stores: per reg, lanes 0-15 / 16-31 each cover a
            // contiguous 32B row segment.
#pragma unroll
            for (int r = 0; r < 8; ++r) {
                float val = acc[mt][r] + bb;
                if (MAT == 0) val *= KSCALE;
                dst[(size_t)(mbase + r + lh * 8) * HEAD + bn] = (_Float16)val;
            }
        } else {
            // V: transpose 16x16 tile in LDS, then one b128 store per lane.
#pragma unroll
            for (int r = 0; r < 8; ++r)
                tile[lid * 16 + (r + lh * 8)] = (_Float16)(acc[mt][r] + bb); // [N][M]
            __syncthreads();   // single-wave workgroup
            const int tn = lane >> 1, tm = (lane & 1) * 8;
            const v4i pk = *(const v4i*)&tile[tn * 16 + tm];
            const int bidx = mbase >> 12;
            const int tt   = (mbase & (TOKENS - 1)) + tm;
            *(v4i*)(dst + ((size_t)bidx * HEAD + n0 + tn) * TOKENS + tt) = pk;
            __syncthreads();
        }
    }
}

// ---------------------------------------------------------------------------
// Stage 2: causal flash-attention, split-s.  grid = (T/16, B), block = 64
// (2 waves).  Wave w handles 32-col chunks c = w, w+2, ... ; private
// (m, l, O) state merged through LDS after the loops.
// wei[t,s] = k[t].q[s]  (reference computes k@q^T), exp2-domain softmax.
// ---------------------------------------------------------------------------
__global__ __launch_bounds__(64) void attn_kernel(
    const _Float16* __restrict__ Kh, const _Float16* __restrict__ Qh,
    const _Float16* __restrict__ Vt, float* __restrict__ out)
{
    __shared__ _Float16 pLds[2 * 16 * 32];  // wave-private P reswizzle slabs
    __shared__ float    oBuf[16 * HEAD];    // wave1 -> wave0 merge buffers
    __shared__ float    mBuf[16], lBuf[16];

    const int tid  = threadIdx.x;
    const int wid  = tid >> 5;
    const int lane = tid & 31;
    const int lh   = lane >> 4;
    const int lid  = lane & 15;
    const int t0   = blockIdx.x * 16;
    const int b    = blockIdx.y;

    const _Float16* Kb = Kh + (size_t)b * TOKENS * HEAD;
    const _Float16* Qb = Qh + (size_t)b * TOKENS * HEAD;
    const _Float16* Vb = Vt + (size_t)b * HEAD * TOKENS;
    _Float16* myP = pLds + wid * (16 * 32);

    // K tile A-fragments (16 rows x 64 h), loaded once.
    const int aoff = lh * 16;
    const _Float16* krow = Kb + (size_t)(t0 + lid) * HEAD;
    const v16h Ka0 = load_frag16(krow + aoff,      krow + aoff + 8);       // h 0..31
    const v16h Ka1 = load_frag16(krow + 32 + aoff, krow + 32 + aoff + 8);  // h 32..63

    float mrow[8], lrow[8];
    v8f O0 = {}, O1 = {}, O2 = {}, O3 = {};
#pragma unroll
    for (int r = 0; r < 8; ++r) { mrow[r] = NEGINF; lrow[r] = 0.0f; }

    const int nchunk = (t0 + 16 + 31) >> 5;
    for (int c = wid; c < nchunk; c += 2) {
        const int s0 = c * 32;

        // ---- S = K_tile @ Q_chunk^T : two 16-col tiles, K-dim 64 ----
        v8f S0 = {}, S1 = {};
        {
            const _Float16* q0 = Qb + (size_t)(s0 + lid) * HEAD + lh * 16;
            const _Float16* q1 = Qb + (size_t)(s0 + 16 + lid) * HEAD + lh * 16;
            v16h B00 = load_frag16(q0,      q0 + 8);
            v16h B01 = load_frag16(q0 + 32, q0 + 40);
            v16h B10 = load_frag16(q1,      q1 + 8);
            v16h B11 = load_frag16(q1 + 32, q1 + 40);
            S0 = __builtin_amdgcn_wmma_f32_16x16x32_f16(false, Ka0, false, B00, (short)0, S0, false, false);
            S0 = __builtin_amdgcn_wmma_f32_16x16x32_f16(false, Ka1, false, B01, (short)0, S0, false, false);
            S1 = __builtin_amdgcn_wmma_f32_16x16x32_f16(false, Ka0, false, B10, (short)0, S1, false, false);
            S1 = __builtin_amdgcn_wmma_f32_16x16x32_f16(false, Ka1, false, B11, (short)0, S1, false, false);
        }

        // ---- online softmax (exp2 domain; scale folded into K) ----
        if (s0 + 31 <= t0) {
            // Fully below the diagonal: no causal mask needed.
#pragma unroll
            for (int r = 0; r < 8; ++r) {
                const float x0 = S0[r], x1 = S1[r];
                float mx = fmaxf(x0, x1);
#pragma unroll
                for (int msk = 8; msk >= 1; msk >>= 1)
                    mx = fmaxf(mx, __shfl_xor(mx, msk, 32));
                const float mnew = fmaxf(mrow[r], mx);
                const float p0 = exp2f(x0 - mnew);
                const float p1 = exp2f(x1 - mnew);
                float ps = p0 + p1;
#pragma unroll
                for (int msk = 8; msk >= 1; msk >>= 1)
                    ps += __shfl_xor(ps, msk, 32);
                const float corr = exp2f(mrow[r] - mnew);
                lrow[r] = lrow[r] * corr + ps;
                mrow[r] = mnew;
                O0[r] *= corr; O1[r] *= corr; O2[r] *= corr; O3[r] *= corr;
                const int M = r + lh * 8;
                myP[M * 32 + lid]      = (_Float16)p0;
                myP[M * 32 + 16 + lid] = (_Float16)p1;
            }
        } else {
            // Diagonal chunk: apply causal mask.
#pragma unroll
            for (int r = 0; r < 8; ++r) {
                const int  trow = t0 + r + lh * 8;
                const bool ok0  = (s0 + lid) <= trow;
                const bool ok1  = (s0 + 16 + lid) <= trow;
                const float x0  = ok0 ? S0[r] : NEGINF;
                const float x1  = ok1 ? S1[r] : NEGINF;
                float mx = fmaxf(x0, x1);
#pragma unroll
                for (int msk = 8; msk >= 1; msk >>= 1)
                    mx = fmaxf(mx, __shfl_xor(mx, msk, 32));
                const float mnew = fmaxf(mrow[r], mx);
                const float p0 = ok0 ? exp2f(x0 - mnew) : 0.0f;
                const float p1 = ok1 ? exp2f(x1 - mnew) : 0.0f;
                float ps = p0 + p1;
#pragma unroll
                for (int msk = 8; msk >= 1; msk >>= 1)
                    ps += __shfl_xor(ps, msk, 32);
                const float corr = exp2f(mrow[r] - mnew);
                lrow[r] = lrow[r] * corr + ps;
                mrow[r] = mnew;
                O0[r] *= corr; O1[r] *= corr; O2[r] *= corr; O3[r] *= corr;
                const int M = r + lh * 8;
                myP[M * 32 + lid]      = (_Float16)p0;
                myP[M * 32 + 16 + lid] = (_Float16)p1;
            }
        }
        lds_order_fence();   // wave-private slab; HW keeps DS ops in order

        // ---- O += P @ V_chunk ----
        const v16h Pf = load_frag16(&myP[lid * 32 + aoff],
                                    &myP[lid * 32 + 8 + aoff]);
        {
            const _Float16* v0 = Vb + (size_t)(0  + lid) * TOKENS + s0 + lh * 16;
            const _Float16* v1 = Vb + (size_t)(16 + lid) * TOKENS + s0 + lh * 16;
            const _Float16* v2 = Vb + (size_t)(32 + lid) * TOKENS + s0 + lh * 16;
            const _Float16* v3 = Vb + (size_t)(48 + lid) * TOKENS + s0 + lh * 16;
            v16h Vf0 = load_frag16(v0, v0 + 8);
            v16h Vf1 = load_frag16(v1, v1 + 8);
            v16h Vf2 = load_frag16(v2, v2 + 8);
            v16h Vf3 = load_frag16(v3, v3 + 8);
            O0 = __builtin_amdgcn_wmma_f32_16x16x32_f16(false, Pf, false, Vf0, (short)0, O0, false, false);
            O1 = __builtin_amdgcn_wmma_f32_16x16x32_f16(false, Pf, false, Vf1, (short)0, O1, false, false);
            O2 = __builtin_amdgcn_wmma_f32_16x16x32_f16(false, Pf, false, Vf2, (short)0, O2, false, false);
            O3 = __builtin_amdgcn_wmma_f32_16x16x32_f16(false, Pf, false, Vf3, (short)0, O3, false, false);
        }
        lds_order_fence();
    }

    // ---- merge the two waves' partial (m, l, O) and store ----
    if (wid == 1) {
#pragma unroll
        for (int r = 0; r < 8; ++r) {
            const int M = r + lh * 8;
            if (lid == 0) { mBuf[M] = mrow[r]; lBuf[M] = lrow[r]; }
            oBuf[M * HEAD +  0 + lid] = O0[r];
            oBuf[M * HEAD + 16 + lid] = O1[r];
            oBuf[M * HEAD + 32 + lid] = O2[r];
            oBuf[M * HEAD + 48 + lid] = O3[r];
        }
    }
    __syncthreads();   // convergent: after both waves' loops
    if (wid == 0) {
#pragma unroll
        for (int r = 0; r < 8; ++r) {
            const int M  = r + lh * 8;
            const float m1 = mBuf[M], l1 = lBuf[M];
            const float mf = fmaxf(mrow[r], m1);
            const float c0 = exp2f(mrow[r] - mf);
            const float c1 = exp2f(m1 - mf);
            const float inv = __builtin_amdgcn_rcpf(lrow[r] * c0 + l1 * c1);
            float* orow = out + ((size_t)b * TOKENS + t0 + M) * HEAD;
            orow[lid]      = (O0[r] * c0 + oBuf[M * HEAD +  0 + lid] * c1) * inv;
            orow[16 + lid] = (O1[r] * c0 + oBuf[M * HEAD + 16 + lid] * c1) * inv;
            orow[32 + lid] = (O2[r] * c0 + oBuf[M * HEAD + 32 + lid] * c1) * inv;
            orow[48 + lid] = (O3[r] * c0 + oBuf[M * HEAD + 48 + lid] * c1) * inv;
        }
    }
}

// ---------------------------------------------------------------------------
extern "C" void kernel_launch(void* const* d_in, const int* in_sizes, int n_in,
                              void* d_out, int out_size, void* d_ws, size_t ws_size,
                              hipStream_t stream) {
    (void)in_sizes; (void)n_in; (void)out_size; (void)ws_size;
    const float* x  = (const float*)d_in[0];
    const float* Wk = (const float*)d_in[1];
    const float* bk = (const float*)d_in[2];
    const float* Wq = (const float*)d_in[3];
    const float* bq = (const float*)d_in[4];
    const float* Wv = (const float*)d_in[5];
    const float* bv = (const float*)d_in[6];
    float* out = (float*)d_out;

    const size_t matBytes = (size_t)BT * HEAD * sizeof(_Float16);  // 2 MB each
    char* ws = (char*)d_ws;
    _Float16* Kh = (_Float16*)(ws);
    _Float16* Qh = (_Float16*)(ws + matBytes);
    _Float16* Vt = (_Float16*)(ws + 2 * matBytes);

    proj_kernel<0><<<dim3(BT / 64, 4), 32, 0, stream>>>(x, Wk, bk, Kh);
    proj_kernel<1><<<dim3(BT / 64, 4), 32, 0, stream>>>(x, Wq, bq, Qh);
    proj_kernel<2><<<dim3(BT / 64, 4), 32, 0, stream>>>(x, Wv, bv, Vt);
    attn_kernel<<<dim3(TOKENS / 16, BATCH), 64, 0, stream>>>(Kh, Qh, Vt, out);
}
